// DenoisingConvNN_2D_K_All_25039659335746
// MI455X (gfx1250) — matrix operand order repaired and checked by reading, not compiled, not run on hardware
//
#include <hip/hip_runtime.h>
#include <hip/hip_bf16.h>

typedef _Float16 v16h __attribute__((ext_vector_type(16)));
typedef _Float16 v8h  __attribute__((ext_vector_type(8)));
typedef float    v8f  __attribute__((ext_vector_type(8)));

#define NBATCH 8
#define NTOK   4096          // 64*64 tokens after pixel_unshuffle(s=2)
#define KNN    9
#define SIMSTRIDE 4100       // 4096 + 4 pad -> rows land on distinct LDS banks

// LDS layout: sim[16][SIMSTRIDE] f32 | candV[16][16][9] f32 | candI[16][16][9] i32 | topIdx[16][9] i32
#define SIM_BYTES   (16 * SIMSTRIDE * 4)
#define CAND_BYTES  (16 * 16 * KNN * 4)
#define SMEM_BYTES  (SIM_BYTES + 2 * CAND_BYTES + 16 * KNN * 4)

// ---------------------------------------------------------------------------
// pixel_unshuffle(x,2) -> token-major xf1[b][n][cc], cc = c*4 + i*2 + j, Cpad=32
// ---------------------------------------------------------------------------
__global__ void prep_unshuffle(const float* __restrict__ x, float* __restrict__ xf1) {
    int t = blockIdx.x * blockDim.x + threadIdx.x;
    if (t >= NBATCH * 12 * NTOK) return;
    int n  = t % NTOK;
    int cc = (t / NTOK) % 12;
    int b  = t / (NTOK * 12);
    int c = cc >> 2, i = (cc >> 1) & 1, j = cc & 1;
    int h = n >> 6, w = n & 63;
    float v = x[(((size_t)b * 3 + c) * 128 + (2 * h + i)) * 128 + (2 * w + j)];
    xf1[((size_t)b * NTOK + n) * 32 + cc] = v;
}

// ---------------------------------------------------------------------------
// per-token L2 normalize (x / max(||x||, 1e-12)), f32 -> f16, zero K padding
// ---------------------------------------------------------------------------
__global__ void normalize_tokens(const float* __restrict__ xf, _Float16* __restrict__ nrm,
                                 int C, int Cpad) {
    int t = blockIdx.x * blockDim.x + threadIdx.x;
    if (t >= NBATCH * NTOK) return;
    const float* src = xf + (size_t)t * Cpad;
    float ss = 0.f;
    for (int c = 0; c < C; ++c) { float v = src[c]; ss += v * v; }
    float inv = 1.0f / fmaxf(sqrtf(ss), 1e-12f);
    _Float16* dst = nrm + (size_t)t * Cpad;
    for (int c = 0; c < C; ++c) dst[c] = (_Float16)(src[c] * inv);
    for (int c = C; c < Cpad; ++c) dst[c] = (_Float16)0.f;
}

// ---------------------------------------------------------------------------
// Fused layer: WMMA Gram stripe (16 x 4096) -> LDS, top-9 per row, gather + conv
// ---------------------------------------------------------------------------
template <int C, int CPAD, int COUT, int COUTPAD, bool RELU>
__global__ __launch_bounds__(256)
void layer_knn_conv(const _Float16* __restrict__ nrm, const float* __restrict__ xf,
                    const float* __restrict__ w, const float* __restrict__ bias,
                    float* __restrict__ out) {
    extern __shared__ __align__(16) char smem[];
    float* sim    = (float*)smem;
    float* candV  = (float*)(smem + SIM_BYTES);
    int*   candI  = (int*)(smem + SIM_BYTES + CAND_BYTES);
    int*   topIdx = (int*)(smem + SIM_BYTES + 2 * CAND_BYTES);

    const int tid  = threadIdx.x;
    const int b    = blockIdx.x >> 8;        // 256 row tiles per batch image
    const int row0 = (blockIdx.x & 255) * 16;

    constexpr int C32 = CPAD / 32;

    // ---- Phase 1: sim[16][4096] = nrm_rows · nrm_colsᵀ via v_wmma_f32_16x16x32_f16.
    // Manual ping-pong software pipeline: each wave owns tiles j=0..31
    // (ct = wave + 8*j); the B-tile for j+1 is in flight during the WMMA on j.
    {
        const int lane = tid & 31;
        const int wave = tid >> 5;
        const int half = lane >> 4;          // K sub-block selector
        const int mrow = lane & 15;          // M (A) / N (B) index
        const int kb   = half * 8;

        // A fragment: 16x32 f16. lane holds row mrow, K = half*16 + i
        const _Float16* arow = nrm + ((size_t)b * NTOK + row0 + mrow) * CPAD;
        v16h afrag[C32];
#pragma unroll
        for (int kt = 0; kt < C32; ++kt) {
            const _Float16* pa = arow + kt * 32;
            afrag[kt] = *(const v16h*)(pa + half * 16);
        }

        // B fragment: 32x16 f16. lane holds column (ct*16+mrow), K = half*16 + i
        const _Float16* bbase = nrm + ((size_t)b * NTOK + mrow) * CPAD + half * 16;

        auto loadB = [&](int j, v16h* frag) {
            const _Float16* p = bbase + (size_t)(wave + 8 * j) * 16 * CPAD;
#pragma unroll
            for (int kt = 0; kt < C32; ++kt)
                frag[kt] = *(const v16h*)(p + kt * 32);
        };
        auto mmaStore = [&](int j, const v16h* frag) {
            v8f acc = {};
#pragma unroll
            for (int kt = 0; kt < C32; ++kt)
                acc = __builtin_amdgcn_wmma_f32_16x16x32_f16(
                    false, afrag[kt], false, frag[kt], (short)0, acc, false, false);
            const int col = (wave + 8 * j) * 16 + mrow;  // C/D: VGPR r -> row r+8*half
#pragma unroll
            for (int r = 0; r < 8; ++r)
                sim[(r + 8 * half) * SIMSTRIDE + col] = acc[r];
        };

        v16h bping[C32], bpong[C32];
        loadB(0, bping);
#pragma unroll 1
        for (int j = 0; j < 30; j += 2) {
            loadB(j + 1, bpong);
            mmaStore(j, bping);              // waits only for bping's loads
            loadB(j + 2, bping);
            mmaStore(j + 1, bpong);          // waits only for bpong's loads
        }
        loadB(31, bpong);
        mmaStore(30, bping);
        mmaStore(31, bpong);
    }
    __syncthreads();

    // ---- Phase 2a: 16 threads per row scan a 256-column strided subset, local top-9.
    // Unrolled x4 so 4 ds_loads are in flight ahead of the insertion chains.
    {
        const int row  = tid & 15;
        const int part = tid >> 4;
        float tv[KNN]; int ti[KNN];
#pragma unroll
        for (int k = 0; k < KNN; ++k) { tv[k] = -2.0f; ti[k] = 0x7fffffff; }
        const float* srow = sim + row * SIMSTRIDE;

        for (int base = part; base < NTOK; base += 64) {
            float vv[4];
#pragma unroll
            for (int u = 0; u < 4; ++u) vv[u] = srow[base + u * 16];
#pragma unroll
            for (int u = 0; u < 4; ++u) {
                const int cidx = base + u * 16;
                const float v = vv[u];
                if (v > tv[KNN - 1] || (v == tv[KNN - 1] && cidx < ti[KNN - 1])) {
                    tv[KNN - 1] = v; ti[KNN - 1] = cidx;
#pragma unroll
                    for (int k = KNN - 1; k > 0; --k) {
                        bool sw = tv[k] > tv[k - 1] ||
                                  (tv[k] == tv[k - 1] && ti[k] < ti[k - 1]);
                        if (sw) {
                            float fv = tv[k]; tv[k] = tv[k - 1]; tv[k - 1] = fv;
                            int   fi = ti[k]; ti[k] = ti[k - 1]; ti[k - 1] = fi;
                        }
                    }
                }
            }
        }
        float* cv = candV + (row * 16 + part) * KNN;
        int*   ci = candI + (row * 16 + part) * KNN;
#pragma unroll
        for (int k = 0; k < KNN; ++k) { cv[k] = tv[k]; ci[k] = ti[k]; }
    }
    __syncthreads();

    // ---- Phase 2b: parallel rank-based merge of 144 candidates per row.
    // All (v,idx) pairs within a row are distinct (parts cover disjoint columns),
    // so ranks are unique and the 9 winners write distinct slots in exact
    // descending-value / lowest-index-tie order (JAX top_k semantics).
    for (int e = tid; e < 16 * 16 * KNN; e += 256) {
        const int row = e / (16 * KNN);
        const int s   = e % (16 * KNN);
        const float* cv = candV + row * 16 * KNN;
        const int*   ci = candI + row * 16 * KNN;
        const float v = cv[s];
        const int   i = ci[s];
        int rank = 0;
        for (int s2 = 0; s2 < 16 * KNN; ++s2) {
            const float v2 = cv[s2];
            const int   i2 = ci[s2];
            rank += (v2 > v || (v2 == v && i2 < i)) ? 1 : 0;
        }
        if (rank < KNN) topIdx[row * KNN + rank] = i;
    }
    __syncthreads();

    // ---- Phase 3a: gather 9 neighbor vectors per row into LDS (reuse sim region)
    constexpr int GSTR = KNN * C + 4;        // bank-spread row stride
    float* gbuf = (float*)smem;              // fits: 16*GSTR*4 <= SIM_BYTES for C<=128
    for (int e = tid; e < 16 * KNN * C; e += 256) {
        int c   = e % C;
        int k   = (e / C) % KNN;
        int row = e / (KNN * C);
        int tok = topIdx[row * KNN + k];
        gbuf[row * GSTR + k * C + c] = xf[((size_t)b * NTOK + tok) * CPAD + c];
    }
    __syncthreads();

    // ---- Phase 3b: out[o,n] = b[o] + sum_{c,k} w[o,c,k] * gathered[c,n,k]  (+ReLU)
    {
        const int row  = tid & 15;
        const int part = tid >> 4;
        const float* g = gbuf + row * GSTR;
        for (int o = part; o < COUT; o += 16) {
            float acc = bias[o];
            const float* wo = w + (size_t)o * C * KNN;
#pragma unroll 1
            for (int c = 0; c < C; ++c) {
                const float* wc = wo + c * KNN;
#pragma unroll
                for (int k = 0; k < KNN; ++k)
                    acc += wc[k] * g[k * C + c];
            }
            if (RELU) acc = fmaxf(acc, 0.f);
            out[((size_t)b * NTOK + row0 + row) * COUTPAD + o] = acc;
        }
    }
}

// ---------------------------------------------------------------------------
// token-major [b][n][cc] (Cpad=16, Cout=12) -> pixel_shuffle -> [8,3,128,128]
// ---------------------------------------------------------------------------
__global__ void shuffle_out(const float* __restrict__ tok, float* __restrict__ out) {
    int t = blockIdx.x * blockDim.x + threadIdx.x;
    if (t >= NBATCH * 3 * 128 * 128) return;
    int W = t & 127;
    int H = (t >> 7) & 127;
    int c = (t >> 14) % 3;
    int b = t / (3 * 128 * 128);
    int h = H >> 1, i = H & 1, w2 = W >> 1, j = W & 1;
    int n  = h * 64 + w2;
    int cc = c * 4 + i * 2 + j;
    out[t] = tok[((size_t)b * NTOK + n) * 16 + cc];
}

// ---------------------------------------------------------------------------
extern "C" void kernel_launch(void* const* d_in, const int* in_sizes, int n_in,
                              void* d_out, int out_size, void* d_ws, size_t ws_size,
                              hipStream_t stream) {
    const float* x  = (const float*)d_in[0];
    const float* w1 = (const float*)d_in[1];
    const float* b1 = (const float*)d_in[2];
    const float* w2 = (const float*)d_in[3];
    const float* b2 = (const float*)d_in[4];
    const float* w3 = (const float*)d_in[5];
    const float* b3 = (const float*)d_in[6];
    float* out = (float*)d_out;

    // workspace carve (~45 MB total)
    char* ws = (char*)d_ws;
    size_t off = 0;
    auto carve = [&](size_t bytes) -> void* {
        void* p = ws + off;
        off = (off + bytes + 255) & ~(size_t)255;
        return p;
    };
    float*    xf1  = (float*)   carve((size_t)NBATCH * NTOK * 32  * 4);
    _Float16* nrm1 = (_Float16*)carve((size_t)NBATCH * NTOK * 32  * 2);
    float*    xf2  = (float*)   carve((size_t)NBATCH * NTOK * 64  * 4);
    _Float16* nrm2 = (_Float16*)carve((size_t)NBATCH * NTOK * 64  * 2);
    float*    xf3  = (float*)   carve((size_t)NBATCH * NTOK * 128 * 4);
    _Float16* nrm3 = (_Float16*)carve((size_t)NBATCH * NTOK * 128 * 2);
    float*    tok  = (float*)   carve((size_t)NBATCH * NTOK * 16  * 4);

    // allow 281 KB dynamic LDS per workgroup (WGP has 320 KB)
    hipFuncSetAttribute(reinterpret_cast<const void*>(layer_knn_conv<12, 32, 64, 64, true>),
                        hipFuncAttributeMaxDynamicSharedMemorySize, SMEM_BYTES);
    hipFuncSetAttribute(reinterpret_cast<const void*>(layer_knn_conv<64, 64, 128, 128, true>),
                        hipFuncAttributeMaxDynamicSharedMemorySize, SMEM_BYTES);
    hipFuncSetAttribute(reinterpret_cast<const void*>(layer_knn_conv<128, 128, 12, 16, false>),
                        hipFuncAttributeMaxDynamicSharedMemorySize, SMEM_BYTES);

    dim3 blk(256);
    dim3 gridLayer(NBATCH * (NTOK / 16));    // 2048 row-blocks

    prep_unshuffle<<<(NBATCH * 12 * NTOK + 255) / 256, blk, 0, stream>>>(x, xf1);
    normalize_tokens<<<(NBATCH * NTOK + 255) / 256, blk, 0, stream>>>(xf1, nrm1, 12, 32);
    layer_knn_conv<12, 32, 64, 64, true><<<gridLayer, blk, SMEM_BYTES, stream>>>(
        nrm1, xf1, w1, b1, xf2);
    normalize_tokens<<<(NBATCH * NTOK + 255) / 256, blk, 0, stream>>>(xf2, nrm2, 64, 64);
    layer_knn_conv<64, 64, 128, 128, true><<<gridLayer, blk, SMEM_BYTES, stream>>>(
        nrm2, xf2, w2, b2, xf3);
    normalize_tokens<<<(NBATCH * NTOK + 255) / 256, blk, 0, stream>>>(xf3, nrm3, 128, 128);
    layer_knn_conv<128, 128, 12, 16, false><<<gridLayer, blk, SMEM_BYTES, stream>>>(
        nrm3, xf3, w3, b3, tok);
    shuffle_out<<<(NBATCH * 3 * 128 * 128 + 255) / 256, blk, 0, stream>>>(tok, out);
}
